// PathwayGAT_38465727103845
// MI455X (gfx1250) — compile-verified, matrix operand order checked by reading, not run on hardware
//
#include <hip/hip_runtime.h>
#include <hip/hip_bf16.h>
#include <stdint.h>

// ---------------- problem constants (match reference) ----------------
#define NN    65536      // nodes
#define BG    32         // graphs
#define NPG   2048       // nodes per graph
#define FIN   128
#define HID   128
#define EE    1048576    // raw edges
#define ETOT  (EE + NN)  // + self loops
#define NCLS  2
#define NEG_SLOPE 0.2f

// ---------------- WMMA types (gfx1250, wave32) ----------------
typedef __attribute__((ext_vector_type(16))) __bf16 v16bf;
typedef __attribute__((ext_vector_type(8)))  __bf16 v8bf;   // 16 bytes
typedef __attribute__((ext_vector_type(8)))  float  v8f;

__device__ __forceinline__ unsigned short f2bfbits(float f) {
    // round-to-nearest-even f32 -> bf16 bit pattern
    unsigned u = __float_as_uint(f);
    u += 0x7FFFu + ((u >> 16) & 1u);
    return (unsigned short)(u >> 16);
}
__device__ __forceinline__ unsigned pack_bf16(float a, float b) {
    return (unsigned)f2bfbits(a) | ((unsigned)f2bfbits(b) << 16);
}

__device__ __forceinline__ void atomicMaxFloat(float* addr, float val) {
    // classic signed/unsigned trick; addr initialized to -inf
    if (val >= 0.f) atomicMax((int*)addr, __float_as_int(val));
    else            atomicMin((unsigned int*)addr, __float_as_uint(val));
}

// =====================================================================
// x (f32) -> xbf (bf16), 4 elements per thread
// =====================================================================
__global__ __launch_bounds__(256) void cvt_bf16(
    const float* __restrict__ in, __bf16* __restrict__ out)
{
    int i = blockIdx.x * 256 + threadIdx.x;     // i in [0, NN*32)
    if (i >= NN * 32) return;
    float4 v = ((const float4*)in)[i];
    uint2 p;
    p.x = pack_bf16(v.x, v.y);
    p.y = pack_bf16(v.z, v.w);
    ((uint2*)out)[i] = p;
}

// =====================================================================
// Wt[n*128 + k] = bf16(W[k*128 + n])   (column-major bf16 weights)
// =====================================================================
__global__ __launch_bounds__(256) void transpose_w(
    const float* __restrict__ W, __bf16* __restrict__ Wt)
{
    int i = blockIdx.x * 256 + threadIdx.x;     // i in [0, 128*128)
    if (i >= 128 * 128) return;
    int n = i >> 7, k = i & 127;
    unsigned short bits = f2bfbits(W[k * 128 + n]);
    ((unsigned short*)Wt)[n * 128 + k] = bits;
}

// =====================================================================
// GEMM: C[N x 128] = A[N x 128] * W[128 x 128]
// A: bf16 row-major in global (fragment = 2x b128 loads, no LDS)
// B: bf16 col-major, staged in LDS with padded stride (2x ds_load_b128)
// block = 256 threads = 8 waves; block tile = 128 rows x 128 cols.
// =====================================================================
#define SWS 136   // padded LDS row stride (bf16 units): 272B -> bank rot 4
__global__ __launch_bounds__(256) void gemm128_wmma(
    const __bf16* __restrict__ Abf, const __bf16* __restrict__ Wt,
    float* __restrict__ C)
{
    __shared__ __align__(16) __bf16 sW[128 * SWS];   // ~34 KB
    const int t = threadIdx.x;
    const size_t rowbase = (size_t)blockIdx.x * 128;

    // stage Wt (col-major) into padded LDS: dword-granular copy
    for (int i = t; i < 128 * 64; i += 256) {        // 64 dwords per column
        int n = i >> 6, kk = i & 63;
        *(uint32_t*)(&sW[n * SWS + kk * 2]) = ((const uint32_t*)Wt)[i];
    }
    __syncthreads();

    const int wave = t >> 5;        // 0..7 : 16-row strip within block
    const int lane = t & 31;
    const int hsel = lane >> 4;     // 0: k 0-7,16-23; 1: k 8-15,24-31 (per 32-k step)
    const int l16  = lane & 15;
    const size_t arow = rowbase + wave * 16 + l16;
    const __bf16* __restrict__ ap = Abf + arow * 128;

    v8f acc[8] = {};                // 8 col-tiles of 16

    #pragma unroll
    for (int kt = 0; kt < 4; ++kt) {                 // K = 4 x 32
        const int kbase = kt * 32 + hsel * 8;
        // A fragment: two contiguous 8-elem (16B) runs: k in [kbase,+8),[kbase+16,+8)
        v8bf a_lo = *(const v8bf*)(ap + kbase);
        v8bf a_hi = *(const v8bf*)(ap + kbase + 16);
        v16bf afrag = __builtin_shufflevector(a_lo, a_hi,
            0,1,2,3,4,5,6,7,8,9,10,11,12,13,14,15);
        #pragma unroll
        for (int n = 0; n < 8; ++n) {                // 8 col tiles, reuse A
            const __bf16* bp = &sW[(n * 16 + l16) * SWS + kbase];
            v8bf b_lo = *(const v8bf*)bp;
            v8bf b_hi = *(const v8bf*)(bp + 16);
            v16bf bfrag = __builtin_shufflevector(b_lo, b_hi,
                0,1,2,3,4,5,6,7,8,9,10,11,12,13,14,15);
            acc[n] = __builtin_amdgcn_wmma_f32_16x16x32_bf16(
                false, afrag, false, bfrag, (short)0, acc[n], false, false);
        }
    }

    // C/D layout: lane(0-15) vgpr v -> M=v, N=lane ; lane(16-31) -> M=v+8
    #pragma unroll
    for (int n = 0; n < 8; ++n) {
        const int colbase = n * 16;
        #pragma unroll
        for (int v = 0; v < 8; ++v) {
            int mrow = v + hsel * 8;
            C[(rowbase + wave * 16 + mrow) * 128 + colbase + l16] = acc[n][v];
        }
    }
}

// =====================================================================
// Per-node attention scores: s_src = h . a_src ; s_dst = h . a_dst
// one wave (32 lanes x float4) per node
// =====================================================================
__global__ __launch_bounds__(256) void scores_kernel(
    const float* __restrict__ h, const float* __restrict__ a_src,
    const float* __restrict__ a_dst, float* __restrict__ s_src,
    float* __restrict__ s_dst)
{
    int g = blockIdx.x * 256 + threadIdx.x;
    int node = g >> 5, lane = g & 31;
    if (node >= NN) return;
    float4 hv = ((const float4*)(h + (size_t)node * 128))[lane];
    float4 as = ((const float4*)a_src)[lane];
    float4 ad = ((const float4*)a_dst)[lane];
    float ps = hv.x * as.x + hv.y * as.y + hv.z * as.z + hv.w * as.w;
    float pd = hv.x * ad.x + hv.y * ad.y + hv.z * ad.z + hv.w * ad.w;
    #pragma unroll
    for (int off = 16; off > 0; off >>= 1) {
        ps += __shfl_down(ps, off, 32);
        pd += __shfl_down(pd, off, 32);
    }
    if (lane == 0) { s_src[node] = ps; s_dst[node] = pd; }
}

// zero aggregation buffer, init segment max/den
__global__ void init_layer(float* __restrict__ agg, float* __restrict__ m,
                           float* __restrict__ den)
{
    int i = blockIdx.x * 256 + threadIdx.x;
    if (i < NN * 128) agg[i] = 0.f;
    if (i < NN) { m[i] = -__builtin_huge_valf(); den[i] = 0.f; }
}

__device__ __forceinline__ void edge_endpoints(const int* __restrict__ ei,
                                               int idx, int& src, int& dst)
{
    if (idx < EE) { src = ei[idx]; dst = ei[EE + idx]; }
    else          { src = dst = idx - EE; }           // self loop
}

// e = leaky_relu(s_src[src] + s_dst[dst]); segment-max into m[dst]
__global__ __launch_bounds__(256) void edge_logits(
    const int* __restrict__ ei, const float* __restrict__ s_src,
    const float* __restrict__ s_dst, float* __restrict__ ebuf,
    float* __restrict__ m)
{
    int idx = blockIdx.x * 256 + threadIdx.x;
    if (idx >= ETOT) return;
    int src, dst; edge_endpoints(ei, idx, src, dst);
    float e = s_src[src] + s_dst[dst];
    e = (e > 0.f) ? e : NEG_SLOPE * e;
    ebuf[idx] = e;
    atomicMaxFloat(m + dst, e);
}

// ex = exp(e - m[dst]); segment-sum into den[dst]
__global__ __launch_bounds__(256) void edge_exp(
    const int* __restrict__ ei, float* __restrict__ ebuf,
    const float* __restrict__ m, float* __restrict__ den)
{
    int idx = blockIdx.x * 256 + threadIdx.x;
    if (idx >= ETOT) return;
    int src, dst; edge_endpoints(ei, idx, src, dst);
    float ex = __expf(ebuf[idx] - m[dst]);
    ebuf[idx] = ex;
    atomicAdd(den + dst, ex);
}

// out[dst] += alpha * h[src]  -- one wave per edge, float4 per lane
__global__ __launch_bounds__(256) void edge_aggregate(
    const int* __restrict__ ei, const float* __restrict__ ebuf,
    const float* __restrict__ den, const float* __restrict__ h,
    float* __restrict__ agg)
{
    int g = blockIdx.x * 256 + threadIdx.x;
    int idx = g >> 5, lane = g & 31;
    if (idx >= ETOT) return;
    int src, dst; edge_endpoints(ei, idx, src, dst);
    float alpha = ebuf[idx] / den[dst];
    float4 hv = ((const float4*)(h + (size_t)src * 128))[lane];
    float* op = agg + (size_t)dst * 128 + lane * 4;
    atomicAdd(op + 0, alpha * hv.x);
    atomicAdd(op + 1, alpha * hv.y);
    atomicAdd(op + 2, alpha * hv.z);
    atomicAdd(op + 3, alpha * hv.w);
}

// agg = relu(agg + b) in place (f32), and emit bf16 copy for next GEMM
__global__ __launch_bounds__(256) void bias_relu_cvt(
    float* __restrict__ agg, const float* __restrict__ b,
    __bf16* __restrict__ outbf)
{
    int i = blockIdx.x * 256 + threadIdx.x;     // i in [0, NN*32), 4 elems each
    if (i >= NN * 32) return;
    float4 v = ((float4*)agg)[i];
    float4 bb = ((const float4*)b)[i & 31];
    v.x = fmaxf(v.x + bb.x, 0.f);
    v.y = fmaxf(v.y + bb.y, 0.f);
    v.z = fmaxf(v.z + bb.z, 0.f);
    v.w = fmaxf(v.w + bb.w, 0.f);
    ((float4*)agg)[i] = v;
    uint2 p;
    p.x = pack_bf16(v.x, v.y);
    p.y = pack_bf16(v.z, v.w);
    ((uint2*)outbf)[i] = p;
}

// r[n] = h2[n] . Wr + br  -- one wave per node
__global__ __launch_bounds__(256) void node_readout(
    const float* __restrict__ h, const float* __restrict__ Wr,
    const float* __restrict__ br, float* __restrict__ r)
{
    int g = blockIdx.x * 256 + threadIdx.x;
    int node = g >> 5, lane = g & 31;
    if (node >= NN) return;
    float4 hv = ((const float4*)(h + (size_t)node * 128))[lane];
    float4 wv = ((const float4*)Wr)[lane];
    float p = hv.x * wv.x + hv.y * wv.y + hv.z * wv.z + hv.w * wv.w;
    #pragma unroll
    for (int off = 16; off > 0; off >>= 1) p += __shfl_down(p, off, 32);
    if (lane == 0) r[node] = p + br[0];
}

// out[b, c] = sum_n r[b*NPG+n] * Wc[n, c] + bc[c]   (32 blocks)
__global__ __launch_bounds__(256) void graph_readout(
    const float* __restrict__ r, const float* __restrict__ Wc,
    const float* __restrict__ bc, float* __restrict__ out)
{
    __shared__ float red0[256];
    __shared__ float red1[256];
    int b = blockIdx.x, t = threadIdx.x;
    float p0 = 0.f, p1 = 0.f;
    for (int n = t; n < NPG; n += 256) {
        float rv = r[(size_t)b * NPG + n];
        p0 += rv * Wc[n * NCLS + 0];
        p1 += rv * Wc[n * NCLS + 1];
    }
    red0[t] = p0; red1[t] = p1;
    __syncthreads();
    for (int s = 128; s > 0; s >>= 1) {
        if (t < s) { red0[t] += red0[t + s]; red1[t] += red1[t + s]; }
        __syncthreads();
    }
    if (t == 0) {
        out[b * NCLS + 0] = red0[0] + bc[0];
        out[b * NCLS + 1] = red1[0] + bc[1];
    }
}

// =====================================================================
extern "C" void kernel_launch(void* const* d_in, const int* in_sizes, int n_in,
                              void* d_out, int out_size, void* d_ws, size_t ws_size,
                              hipStream_t stream)
{
    (void)in_sizes; (void)n_in; (void)out_size; (void)ws_size;
    const float* x   = (const float*)d_in[0];
    const int*   ei  = (const int*)  d_in[1];
    // d_in[2] = batch: implicit via equal-size graph reshape, unused
    const float* W1  = (const float*)d_in[3];
    const float* as1 = (const float*)d_in[4];
    const float* ad1 = (const float*)d_in[5];
    const float* b1  = (const float*)d_in[6];
    const float* W2  = (const float*)d_in[7];
    const float* as2 = (const float*)d_in[8];
    const float* ad2 = (const float*)d_in[9];
    const float* b2  = (const float*)d_in[10];
    const float* Wr  = (const float*)d_in[11];
    const float* br  = (const float*)d_in[12];
    const float* Wc  = (const float*)d_in[13];
    const float* bc  = (const float*)d_in[14];
    float* out = (float*)d_out;

    // workspace layout
    float* ws    = (float*)d_ws;
    float* h     = ws;                          // N*128 f32
    float* agg   = h   + (size_t)NN * 128;      // N*128 f32
    float* s_src = agg + (size_t)NN * 128;      // N
    float* s_dst = s_src + NN;                  // N
    float* m     = s_dst + NN;                  // N
    float* den   = m     + NN;                  // N
    float* r     = den   + NN;                  // N
    float* ebuf  = r     + NN;                  // ETOT
    __bf16* xbf  = (__bf16*)(ebuf + ETOT);      // N*128 bf16 (GEMM A operand)
    __bf16* Wt   = xbf + (size_t)NN * 128;      // 128*128 bf16 (col-major W)

    const int gemmBlocks  = NN / 128;           // 512
    const int nodeWaveBlk = NN / 8;             // 8192 (wave per node)
    const int elemBlocks  = (NN * 128) / 256;   // 32768
    const int vec4Blocks  = (NN * 32) / 256;    // 8192  (float4-granular)
    const int edgeBlocks  = (ETOT + 255) / 256; // 4352
    const int edgeWaveBlk = (ETOT * 32) / 256;  // 139264 (wave per edge)

    auto gat_layer = [&](const float* W, const float* as,
                         const float* ad, const float* b) {
        transpose_w   <<<64,          256, 0, stream>>>(W, Wt);
        gemm128_wmma  <<<gemmBlocks,  256, 0, stream>>>(xbf, Wt, h);
        scores_kernel <<<nodeWaveBlk, 256, 0, stream>>>(h, as, ad, s_src, s_dst);
        init_layer    <<<elemBlocks,  256, 0, stream>>>(agg, m, den);
        edge_logits   <<<edgeBlocks,  256, 0, stream>>>(ei, s_src, s_dst, ebuf, m);
        edge_exp      <<<edgeBlocks,  256, 0, stream>>>(ei, ebuf, m, den);
        edge_aggregate<<<edgeWaveBlk, 256, 0, stream>>>(ei, ebuf, den, h, agg);
        bias_relu_cvt <<<vec4Blocks,  256, 0, stream>>>(agg, b, xbf);
    };

    cvt_bf16<<<vec4Blocks, 256, 0, stream>>>(x, xbf);   // xbf = bf16(x)
    gat_layer(W1, as1, ad1, b1);    // agg = relu(GAT(x));  xbf = bf16(agg)
    gat_layer(W2, as2, ad2, b2);    // agg = relu(GAT(agg))

    node_readout <<<nodeWaveBlk, 256, 0, stream>>>(agg, Wr, br, r);
    graph_readout<<<BG,          256, 0, stream>>>(r, Wc, bc, out);
}
